// FuserBlock_62895501082939
// MI455X (gfx1250) — compile-verified, matrix operand order checked by reading, not run on hardware
//
#include <hip/hip_runtime.h>
#include <math.h>

// ---------------- types ----------------
typedef _Float16 v16h __attribute__((ext_vector_type(16)));
typedef _Float16 v8h  __attribute__((ext_vector_type(8)));
typedef float    v8f  __attribute__((ext_vector_type(8)));
typedef float    v4f  __attribute__((ext_vector_type(4)));

#define INV_SQRT_DF 0.17677669529663687f

// ---------------- workspace layout (float offsets) ----------------
#define OFF_Q     0
#define OFF_K     262144
#define OFF_V     524288
#define OFF_WV    786432        // weighted_V * (1+yx2)
#define OFF_YE2   1048576       // 4*256
#define OFF_YX2   1049600       // 4*256
#define OFF_XS    1050624       // 4*1024
#define OFF_EPART 1054720       // 64*64*4
#define OFF_F16   1071104       // half region starts here
// half offsets inside half region (packed B-fragment weights)
#define H_Q      0
#define H_K      65536
#define H_V      131072
#define H_EMUL   196608
#define H_XOUT   212992
#define H_EOUTB  278528         // 4 batches * 16384 (W_e_out scaled by (1+ye2[b]))
#define H_TOTAL  344064

#define YSTR 264                 // padded LDS row stride (halves) for bank-conflict-free TR reads

// ---------------- WMMA helpers ----------------
static __device__ __forceinline__ v8f wmma16(v16h a, v16h b, v8f c) {
  // D = A(16x32 f16) * B(32x16 f16) + C(16x16 f32)
  return __builtin_amdgcn_wmma_f32_16x16x32_f16(false, a, false, b, (short)0, c, false, false);
}

// Build an A fragment from an f32 row. p points at (row m_lane, k = kf*32 + kb).
// Per ISA layout: lane holds k = kb+{0..7} then kb+16+{0..7}  (kb = 8 for lanes >= 16).
static __device__ __forceinline__ v16h load_a_f32(const float* __restrict__ p) {
  v4f f0 = *(const v4f*)(p);
  v4f f1 = *(const v4f*)(p + 4);
  v4f f2 = *(const v4f*)(p + 16);
  v4f f3 = *(const v4f*)(p + 20);
  v16h a;
#pragma unroll
  for (int i = 0; i < 4; ++i) {
    a[i]      = (_Float16)f0[i];
    a[4 + i]  = (_Float16)f1[i];
    a[8 + i]  = (_Float16)f2[i];
    a[12 + i] = (_Float16)f3[i];
  }
  return a;
}

// Same A-fragment pattern but from f16 LDS.
static __device__ __forceinline__ v16h load_a_lds(const _Float16* p) {
  v8h lo = *(const v8h*)(p);
  v8h hi = *(const v8h*)(p + 16);
  v16h a;
#pragma unroll
  for (int i = 0; i < 8; ++i) { a[i] = lo[i]; a[8 + i] = hi[i]; }
  return a;
}

// ---------------- P1: ye2 = y@W_y_e_mul+b, yx2 = y@W_y_x_mul+b ----------------
__global__ void __launch_bounds__(256) k_pre1(const float* __restrict__ y,
    const float* __restrict__ Wye, const float* __restrict__ bye,
    const float* __restrict__ Wyx, const float* __restrict__ byx,
    float* __restrict__ ye2, float* __restrict__ yx2) {
  const int d = threadIdx.x;
  for (int b = 0; b < 4; ++b) {
    float ae = bye[d], ax = byx[d];
    for (int k = 0; k < 64; ++k) {
      const float yv = y[b * 64 + k];
      ae += yv * Wye[k * 256 + d];
      ax += yv * Wyx[k * 256 + d];
    }
    ye2[b * 256 + d] = ae;
    yx2[b * 256 + d] = ax;
  }
}

// ---------------- P2: pack weights to f16 B-fragment order ----------------
// packed[((nt*KF + kf)*32 + lane)*16 + idx] = W[k*N + n]
//   k = kf*32 + (lane&16?8:0) + (idx&7) + ((idx>>3)<<4),  n = nt*16 + (lane&15)
__global__ void __launch_bounds__(256) k_pack(const float* __restrict__ Wq,
    const float* __restrict__ Wk, const float* __restrict__ Wv,
    const float* __restrict__ Wemul, const float* __restrict__ Wxout,
    const float* __restrict__ Weout, const float* __restrict__ ye2,
    _Float16* __restrict__ dst) {
  const int e = blockIdx.x * 256 + threadIdx.x;
  if (e >= H_TOTAL) return;
  const float* W = Wq;
  int Nd = 256, KF = 8, rel = e, bsel = -1;
  if (e < H_K)           { W = Wq;    rel = e;           }
  else if (e < H_V)      { W = Wk;    rel = e - H_K;     }
  else if (e < H_EMUL)   { W = Wv;    rel = e - H_V;     }
  else if (e < H_XOUT)   { W = Wemul; rel = e - H_EMUL;  KF = 2; }
  else if (e < H_EOUTB)  { W = Wxout; rel = e - H_XOUT;  }
  else { rel = e - H_EOUTB; bsel = rel >> 14; rel &= 16383; W = Weout; Nd = 64; }
  const int idx  = rel & 15;
  const int lane = (rel >> 4) & 31;
  const int tt   = rel >> 9;
  const int kf   = tt % KF;
  const int nt   = tt / KF;
  const int kbl  = (lane & 16) ? 8 : 0;
  const int k    = kf * 32 + kbl + (idx & 7) + ((idx >> 3) << 4);
  const int n    = nt * 16 + (lane & 15);
  float v = W[k * Nd + n];
  if (bsel >= 0) v *= (1.0f + ye2[bsel * 256 + k]);  // fold (1+ye2[b,k]) into W_e_out
  dst[e] = (_Float16)v;
}

// ---------------- row GEMM: out[r,:] = (A[r,:]@W + bias) * rowmask[r] * xs ----------------
// A: 1024x256 f32, W packed (NT=16, KF=8), out: 1024x256 f32. Used for Q,K,V and x_out.
__global__ void __launch_bounds__(256) k_gemm_rows(const float* __restrict__ A,
    const _Float16* __restrict__ pW, const float* __restrict__ bias,
    const float* __restrict__ rowmask, float extrascale, float* __restrict__ out) {
  const int lane = threadIdx.x & 31;
  const int wv = threadIdx.x >> 5;
  const int m = lane & 15;
  const int kb = (lane & 16) ? 8 : 0;
  const int mtile = blockIdx.x;
  const float* ap = A + (mtile * 16 + m) * 256;
  v16h af[8];
#pragma unroll
  for (int kf = 0; kf < 8; ++kf) af[kf] = load_a_f32(ap + kf * 32 + kb);
#pragma unroll
  for (int q = 0; q < 2; ++q) {
    const int nt = wv * 2 + q;
    const int n = nt * 16 + m;
    const float bv = bias[n];
    v8f c;
#pragma unroll
    for (int v = 0; v < 8; ++v) c[v] = bv;
#pragma unroll
    for (int kf = 0; kf < 8; ++kf) {
      v16h bfr = *(const v16h*)(pW + ((nt * 8 + kf) * 32 + lane) * 16);
      c = wmma16(af[kf], bfr, c);
    }
#pragma unroll
    for (int v = 0; v < 8; ++v) {
      const int r = mtile * 16 + v + ((lane >> 4) << 3);
      out[r * 256 + n] = c[v] * rowmask[r] * extrascale;
    }
  }
}

// ---------------- main fused kernel: one workgroup per (b,i) ----------------
// E1 = E@W_e_mul (WMMA) -> Y = Q*K*(E1+1) into LDS (f16, padded rows)
// newE = Y @ (W_e_out * (1+ye2[b])) (WMMA from LDS) -> HBM
// softmax over j per channel + attn*V reduction -> wvs
__global__ void __launch_bounds__(256) k_main(const float* __restrict__ E,
    const float* __restrict__ mask, const float* __restrict__ Qs,
    const float* __restrict__ Ks, const float* __restrict__ Vs,
    const float* __restrict__ b_e_mul, const float* __restrict__ b_e_out,
    const _Float16* __restrict__ pWem, const _Float16* __restrict__ pWeo,
    const float* __restrict__ yx2, float* __restrict__ outE,
    float* __restrict__ wvs) {
  extern __shared__ _Float16 Ylds[];  // 256 rows x YSTR halves
  const int lane = threadIdx.x & 31;
  const int wv = threadIdx.x >> 5;
  const int m = lane & 15;
  const int kb = (lane & 16) ? 8 : 0;
  const int hi8 = (lane >> 4) << 3;
  const int b = blockIdx.x >> 8;
  const int i = blockIdx.x & 255;
  const float mi = mask[b * 256 + i];
  const float* qrow = Qs + (b * 256 + i) * 256;  // Q already masked & /sqrt(DF)

  // prefetch second j-tile's E rows while working on the first
  __builtin_prefetch(E + ((b * 256 + i) * 256 + (wv + 8) * 16 + m) * 64, 0, 0);

  // ---- phase A: E1 GEMM + Y into LDS ----
#pragma unroll
  for (int t = 0; t < 2; ++t) {
    const int jb = (wv + t * 8) * 16;
    const float* erow = E + ((b * 256 + i) * 256 + jb + m) * 64;
    const v16h ea0 = load_a_f32(erow + kb);        // K = 0..31
    const v16h ea1 = load_a_f32(erow + 32 + kb);   // K = 32..63
    float mj[8];
#pragma unroll
    for (int v = 0; v < 8; ++v) mj[v] = mask[b * 256 + jb + v + hi8];
    for (int nt = 0; nt < 16; ++nt) {
      const int d = nt * 16 + m;
      const float bv = b_e_mul[d];
      v8f c;
#pragma unroll
      for (int v = 0; v < 8; ++v) c[v] = bv;
      const v16h b0 = *(const v16h*)(pWem + ((nt * 2 + 0) * 32 + lane) * 16);
      const v16h b1 = *(const v16h*)(pWem + ((nt * 2 + 1) * 32 + lane) * 16);
      c = wmma16(ea0, b0, c);
      c = wmma16(ea1, b1, c);
      const float qv = qrow[d];
#pragma unroll
      for (int v = 0; v < 8; ++v) {
        const int jm = jb + v + hi8;
        const float kv = Ks[(b * 256 + jm) * 256 + d];
        const float yv = qv * kv * (c[v] * mi * mj[v] + 1.0f);
        Ylds[jm * YSTR + d] = (_Float16)yv;
      }
    }
  }
  __syncthreads();

  // ---- phase B: newE = Y @ W_e_out' + b, masked ----
#pragma unroll
  for (int t = 0; t < 2; ++t) {
    const int jb = (wv + t * 8) * 16;
    const _Float16* yrow = Ylds + (jb + m) * YSTR;
    float mj[8];
#pragma unroll
    for (int v = 0; v < 8; ++v) mj[v] = mask[b * 256 + jb + v + hi8];
#pragma unroll
    for (int nt = 0; nt < 4; ++nt) {
      const int n = nt * 16 + m;
      const float bv = b_e_out[n];
      v8f c;
#pragma unroll
      for (int v = 0; v < 8; ++v) c[v] = bv;
#pragma unroll
      for (int kf = 0; kf < 8; ++kf) {
        const v16h a  = load_a_lds(yrow + kf * 32 + kb);
        const v16h bf = *(const v16h*)(pWeo + b * 16384 + ((nt * 8 + kf) * 32 + lane) * 16);
        c = wmma16(a, bf, c);
      }
#pragma unroll
      for (int v = 0; v < 8; ++v) {
        const int jm = jb + v + hi8;
        outE[((b * 256 + i) * 256 + jm) * 64 + n] = c[v] * mi * mj[v];
      }
    }
  }

  // ---- phase C: per-channel softmax over j + attn*V ----
  const int d = wv * 32 + lane;  // 256 channels across 8 waves
  float mmax = -3.0e38f;
  for (int j = 0; j < 256; ++j)
    if (mask[b * 256 + j] > 0.0f)
      mmax = fmaxf(mmax, (float)Ylds[j * YSTR + d]);
  float s = 0.0f, acc = 0.0f;
  const float* vcol = Vs + b * 65536 + d;
  for (int j = 0; j < 256; ++j) {
    if (mask[b * 256 + j] > 0.0f) {
      const float e = __expf((float)Ylds[j * YSTR + d] - mmax);
      s += e;
      acc += e * vcol[j * 256];
    }
  }
  wvs[(b * 256 + i) * 256 + d] =
      (acc / fmaxf(s, 1e-30f)) * (1.0f + yx2[b * 256 + d]);
}

// ---------------- X pooling stats ----------------
__global__ void __launch_bounds__(256) k_xstats(const float* __restrict__ X,
                                                float* __restrict__ xs) {
  const int b = blockIdx.x, d = threadIdx.x;
  float sum = 0.f, sq = 0.f, mn = 3e38f, mx = -3e38f;
  for (int i = 0; i < 256; ++i) {
    const float v = X[(b * 256 + i) * 256 + d];
    sum += v; sq += v * v; mn = fminf(mn, v); mx = fmaxf(mx, v);
  }
  const float mean = sum * (1.0f / 256.0f);
  const float var = (sq - 256.0f * mean * mean) * (1.0f / 255.0f);
  xs[b * 1024 + d] = mean;
  xs[b * 1024 + 256 + d] = mn;
  xs[b * 1024 + 512 + d] = mx;
  xs[b * 1024 + 768 + d] = sqrtf(fmaxf(var, 0.0f));
}

// ---------------- E pooling stats, stage 1 (partials; E is L2-resident) ----------------
__global__ void __launch_bounds__(256) k_epart(const float* __restrict__ E,
                                               float* __restrict__ ep) {
  const int b = blockIdx.x >> 4, s = blockIdx.x & 15;
  const int de = threadIdx.x & 63, g = threadIdx.x >> 6;
  float sum = 0.f, sq = 0.f, mn = 3e38f, mx = -3e38f;
  const float* base = E + ((size_t)b * 65536 + s * 4096 + g * 1024) * 64 + de;
  for (int r = 0; r < 1024; ++r) {
    const float v = base[(size_t)r * 64];
    sum += v; sq += v * v; mn = fminf(mn, v); mx = fmaxf(mx, v);
  }
  __shared__ float red[256 * 4];
  red[threadIdx.x * 4 + 0] = sum;
  red[threadIdx.x * 4 + 1] = sq;
  red[threadIdx.x * 4 + 2] = mn;
  red[threadIdx.x * 4 + 3] = mx;
  __syncthreads();
  if (threadIdx.x < 64) {
    const int dd = threadIdx.x;
    float S = 0.f, Q = 0.f, MN = 3e38f, MX = -3e38f;
    for (int gg = 0; gg < 4; ++gg) {
      const float* p = red + (gg * 64 + dd) * 4;
      S += p[0]; Q += p[1]; MN = fminf(MN, p[2]); MX = fmaxf(MX, p[3]);
    }
    float* o = ep + ((b * 16 + s) * 64 + dd) * 4;
    o[0] = S; o[1] = Q; o[2] = MN; o[3] = MX;
  }
}

// ---------------- y path: reduce E stats + tiny MLP ----------------
__global__ void __launch_bounds__(256) k_ypath(const float* __restrict__ ep,
    const float* __restrict__ xs, const float* __restrict__ y,
    const float* __restrict__ Wyy, const float* __restrict__ byy,
    const float* __restrict__ Wxy, const float* __restrict__ bxy,
    const float* __restrict__ Wey, const float* __restrict__ bey,
    const float* __restrict__ W1, const float* __restrict__ b1,
    const float* __restrict__ W2, const float* __restrict__ b2,
    float* __restrict__ outY) {
  __shared__ float es[1024];
  __shared__ float t1[256];
  __shared__ float hh[256];
  const int t = threadIdx.x, b = t >> 6, c = t & 63;
  {
    float S = 0.f, Q = 0.f, MN = 3e38f, MX = -3e38f;
    for (int s = 0; s < 16; ++s) {
      const float* p = ep + ((b * 16 + s) * 64 + c) * 4;
      S += p[0]; Q += p[1]; MN = fminf(MN, p[2]); MX = fmaxf(MX, p[3]);
    }
    const float mean = S * (1.0f / 65536.0f);
    const float var = (Q - 65536.0f * mean * mean) * (1.0f / 65535.0f);
    es[b * 256 + c] = mean;
    es[b * 256 + 64 + c] = MN;
    es[b * 256 + 128 + c] = MX;
    es[b * 256 + 192 + c] = sqrtf(fmaxf(var, 0.0f));
  }
  __syncthreads();
  {
    float a = byy[c] + bxy[c] + bey[c];
    for (int k = 0; k < 64; ++k)   a += y[b * 64 + k] * Wyy[k * 64 + c];
    for (int k = 0; k < 1024; ++k) a += xs[b * 1024 + k] * Wxy[k * 64 + c];
    for (int k = 0; k < 256; ++k)  a += es[b * 256 + k] * Wey[k * 64 + c];
    t1[t] = a;
  }
  __syncthreads();
  {
    float a = b1[c];
    for (int k = 0; k < 64; ++k) a += t1[b * 64 + k] * W1[k * 64 + c];
    hh[t] = fmaxf(a, 0.0f);
  }
  __syncthreads();
  {
    float a = b2[c];
    for (int k = 0; k < 64; ++k) a += hh[b * 64 + k] * W2[k * 64 + c];
    outY[t] = a;
  }
}

// ---------------- launch ----------------
extern "C" void kernel_launch(void* const* d_in, const int* in_sizes, int n_in,
                              void* d_out, int out_size, void* d_ws, size_t ws_size,
                              hipStream_t stream) {
  (void)in_sizes; (void)n_in; (void)out_size; (void)ws_size;
  const float* X         = (const float*)d_in[0];
  const float* E         = (const float*)d_in[1];
  const float* y         = (const float*)d_in[2];
  const float* nm        = (const float*)d_in[3];
  const float* W_q       = (const float*)d_in[4];
  const float* b_q       = (const float*)d_in[5];
  const float* W_k       = (const float*)d_in[6];
  const float* b_k       = (const float*)d_in[7];
  const float* W_v       = (const float*)d_in[8];
  const float* b_v       = (const float*)d_in[9];
  const float* W_e_mul   = (const float*)d_in[10];
  const float* b_e_mul   = (const float*)d_in[11];
  const float* W_y_e_mul = (const float*)d_in[12];
  const float* b_y_e_mul = (const float*)d_in[13];
  const float* W_y_x_mul = (const float*)d_in[14];
  const float* b_y_x_mul = (const float*)d_in[15];
  const float* W_y_y     = (const float*)d_in[16];
  const float* b_y_y     = (const float*)d_in[17];
  const float* W_x_y     = (const float*)d_in[18];
  const float* b_x_y     = (const float*)d_in[19];
  const float* W_e_y     = (const float*)d_in[20];
  const float* b_e_y     = (const float*)d_in[21];
  const float* W_x_out   = (const float*)d_in[22];
  const float* b_x_out   = (const float*)d_in[23];
  const float* W_e_out   = (const float*)d_in[24];
  const float* b_e_out   = (const float*)d_in[25];
  const float* W_y_out1  = (const float*)d_in[26];
  const float* b_y_out1  = (const float*)d_in[27];
  const float* W_y_out2  = (const float*)d_in[28];
  const float* b_y_out2  = (const float*)d_in[29];

  float* ws = (float*)d_ws;
  _Float16* wh = (_Float16*)(ws + OFF_F16);
  float* outX = (float*)d_out;            // (4,256,256)
  float* outE = outX + 262144;            // (4,256,256,64)
  float* outY = outX + 17039360;          // (4,64)

  k_pre1<<<1, 256, 0, stream>>>(y, W_y_e_mul, b_y_e_mul, W_y_x_mul, b_y_x_mul,
                                ws + OFF_YE2, ws + OFF_YX2);
  k_pack<<<(H_TOTAL + 255) / 256, 256, 0, stream>>>(W_q, W_k, W_v, W_e_mul,
                                                    W_x_out, W_e_out,
                                                    ws + OFF_YE2, wh);
  k_gemm_rows<<<64, 256, 0, stream>>>(X, wh + H_Q, b_q, nm, INV_SQRT_DF, ws + OFF_Q);
  k_gemm_rows<<<64, 256, 0, stream>>>(X, wh + H_K, b_k, nm, 1.0f, ws + OFF_K);
  k_gemm_rows<<<64, 256, 0, stream>>>(X, wh + H_V, b_v, nm, 1.0f, ws + OFF_V);
  k_main<<<1024, 256, 256 * YSTR * sizeof(_Float16), stream>>>(
      E, nm, ws + OFF_Q, ws + OFF_K, ws + OFF_V, b_e_mul, b_e_out,
      wh + H_EMUL, wh + H_EOUTB, ws + OFF_YX2, outE, ws + OFF_WV);
  k_gemm_rows<<<64, 256, 0, stream>>>(ws + OFF_WV, wh + H_XOUT, b_x_out, nm, 1.0f, outX);
  k_xstats<<<4, 256, 0, stream>>>(X, ws + OFF_XS);
  k_epart<<<64, 256, 0, stream>>>(E, ws + OFF_EPART);
  k_ypath<<<1, 256, 0, stream>>>(ws + OFF_EPART, ws + OFF_XS, y, W_y_y, b_y_y,
                                 W_x_y, b_x_y, W_e_y, b_e_y, W_y_out1, b_y_out1,
                                 W_y_out2, b_y_out2, outY);
}
